// MultiHeadAttention_45947560133060
// MI455X (gfx1250) — matrix operand order repaired and checked by reading, not compiled, not run on hardware
//
#include <hip/hip_runtime.h>
#include <math.h>

// ---- fixed problem dims (match reference) ----
#define D_MODEL 1024
#define SEQ     2048
#define BATCH   2
#define NH      16
#define DK      64
#define MROWS   (BATCH * SEQ)   // 4096

typedef __attribute__((ext_vector_type(16))) _Float16 v16h;
typedef __attribute__((ext_vector_type(8)))  _Float16 v8h;
typedef __attribute__((ext_vector_type(4)))  _Float16 v4h;
typedef __attribute__((ext_vector_type(8)))  float    v8f;

// Tensor-DMA availability (device pass only; fallback = manual staging)
#if defined(__HIP_DEVICE_COMPILE__) && defined(__gfx1250__) && \
    __has_builtin(__builtin_amdgcn_tensor_load_to_lds) &&      \
    __has_builtin(__builtin_amdgcn_s_wait_tensorcnt)
#define USE_TDM 1
#else
#define USE_TDM 0
#endif

// ---------------------------------------------------------------------------
// WMMA helpers (CDNA5 wave32, 16x16x32 f16 -> f32)
// A fragment (16x32, MxK): lane l -> row M = l&15; ksel = l>>4.
//   v16h elems 0..7  = K (k0 + ksel*8 + 0..7)
//   v16h elems 8..15 = K (k0 + 16 + ksel*8 + 0..7)
// B fragment (32x16, KxN): lane l -> col N = l&15; khalf = l>>4.
//   v16h elems 0..7 = K (k0 + khalf*8 + 0..7), 8..15 = K (k0 + 16 + khalf*8 + 0..7).
// C/D (16x16 f32): VGPR r, lane l -> row M = r + 8*(l>>4), col N = l&15.
// ---------------------------------------------------------------------------

static __device__ __forceinline__ v8f wmma_f16(v16h a, v16h b, v8f c) {
  return __builtin_amdgcn_wmma_f32_16x16x32_f16(false, a, false, b, (short)0, c,
                                                false, false);
}

union frag16 { v16h v; v8h h[2]; };

static __device__ __forceinline__ v16h load_a_frag(const _Float16* __restrict__ A,
                                                   int lda, int row0, int k0, int lane) {
  const int m = lane & 15, ksel = lane >> 4;
  const _Float16* p = A + (size_t)(row0 + m) * lda + k0 + ksel * 8;
  frag16 u;
  u.h[0] = *(const v8h*)(p);        // K offsets +0..7
  u.h[1] = *(const v8h*)(p + 16);   // K offsets +16..23
  return u.v;
}

// B fragment from a row-major [N,K] matrix (B[k][n] = W[n][k], i.e. X @ W^T).
static __device__ __forceinline__ v16h load_b_frag_wT(const _Float16* __restrict__ W,
                                                      int ldw, int col0, int k0, int lane) {
  const int n = lane & 15, khalf = lane >> 4;
  const _Float16* p = W + (size_t)(col0 + n) * ldw + k0 + khalf * 8;
  frag16 u;
  u.h[0] = *(const v8h*)(p);
  u.h[1] = *(const v8h*)(p + 16);
  return u.v;
}

// ---------------------------------------------------------------------------
// Tensor Data Mover: async 2D tile (rows x DK f16) global -> LDS, with HW
// padding of 4 DWORDs per 32-DWORD row -> LDS row pitch of 72 halves (= LDK).
// ---------------------------------------------------------------------------
#if USE_TDM
static __device__ __forceinline__ void tdm_load_kv_tile(const _Float16* gsrc,
                                                        unsigned lds_byte_addr,
                                                        int rows_remaining) {
  typedef unsigned int u32x4 __attribute__((ext_vector_type(4)));
  typedef int i32x8 __attribute__((ext_vector_type(8)));
  typedef int i32x4 __attribute__((ext_vector_type(4)));

  const unsigned long long ga = (unsigned long long)(uintptr_t)gsrc;

  u32x4 g0;
  g0[0] = 1u;                                      // count=1, user descriptor
  g0[1] = lds_byte_addr;                           // lds_addr
  g0[2] = (unsigned)(ga & 0xFFFFFFFFull);          // global_addr[31:0]
  g0[3] = (unsigned)((ga >> 32) & 0x01FFFFFFull)   // global_addr[56:32]
          | (2u << 30);                            // type = 2 ("image")

  const unsigned rows = (unsigned)rows_remaining;
  i32x8 g1;
  // data_size=2B (code 1), pad_enable, pad_interval=32 DWORDs (code 4),
  // pad_amount=4 DWORDs (code 3); workgroup_mask=0 (not in a cluster).
  g1[0] = (int)((1u << 16) | (1u << 20) | (4u << 22) | (3u << 25));
  g1[1] = (int)(((unsigned)DK & 0xFFFFu) << 16);   // tensor_dim0[15:0] @ bits 63:48
  g1[2] = (int)((rows & 0xFFFFu) << 16);           // dim0 hi16 (=0) | tensor_dim1 lo16
  g1[3] = (int)(((rows >> 16) & 0xFFFFu)           // tensor_dim1 hi16
                | ((unsigned)DK << 16));           // tile_dim0 = 64
  g1[4] = 64;                                      // tile_dim1 = 64 (tile_dim2 = 0)
  g1[5] = DK;                                      // tensor_dim0_stride = 64 elems
  g1[6] = 0;
  g1[7] = 0;

  i32x4 z4 = {0, 0, 0, 0};
#if __clang_major__ >= 23
  i32x8 z8 = {0, 0, 0, 0, 0, 0, 0, 0};
  __builtin_amdgcn_tensor_load_to_lds(g0, g1, z4, z4, z8, 0);
#else
  __builtin_amdgcn_tensor_load_to_lds(g0, g1, z4, z4, 0);
#endif
}
#endif

// ---------------------------------------------------------------------------
// f32 -> f16 cast (n must be a multiple of 4; all our sizes are)
// ---------------------------------------------------------------------------
__global__ __launch_bounds__(256) void cast_f32_f16(const float* __restrict__ in,
                                                    _Float16* __restrict__ out, int n) {
  int i = (blockIdx.x * 256 + threadIdx.x) * 4;
  if (i + 3 < n) {
    float4 f = *(const float4*)(in + i);
    v4h h;
    h[0] = (_Float16)f.x; h[1] = (_Float16)f.y;
    h[2] = (_Float16)f.z; h[3] = (_Float16)f.w;
    *(v4h*)(out + i) = h;
  }
}

// ---------------------------------------------------------------------------
// GEMM: C[M,N] = A[M,K] @ W[N,K]^T + bias[N]
// Block = 256 threads = 8 waves; wave computes a 16x128 tile (8 WMMA tiles,
// A-fragment amortized over 8 WMMAs per k-step).
// Block tile = 32(M) x 512(N). Grid = (N/512, M/32).
// OUT_F32:  write f32 to outF [M,N] row-major (final projection -> d_out)
// HEAD_SPLIT: write f16 to outH laid out [B, H, S, DK] (QKV projections)
// ---------------------------------------------------------------------------
template <bool OUT_F32, bool HEAD_SPLIT>
__global__ __launch_bounds__(256) void gemm_xwT(const _Float16* __restrict__ A,
                                                const _Float16* __restrict__ W,
                                                const float* __restrict__ bias,
                                                float* __restrict__ outF,
                                                _Float16* __restrict__ outH,
                                                int K) {
  const int lane = threadIdx.x & 31;
  const int wave = threadIdx.x >> 5;
  const int tileM = blockIdx.y * 32 + (wave & 1) * 16;
  const int tileN = blockIdx.x * 512 + (wave >> 1) * 128;

  v8f acc[8] = {};

  for (int k0 = 0; k0 < K; k0 += 32) {
    if (k0 + 64 < K) {
      // hint the next A chunk into cache (global_prefetch_b8)
      __builtin_prefetch(A + (size_t)(tileM + (lane & 15)) * K + k0 + 64, 0, 1);
    }
    v16h a = load_a_frag(A, K, tileM, k0, lane);
#pragma unroll
    for (int t = 0; t < 8; ++t) {
      v16h b = load_b_frag_wT(W, K, tileN + t * 16, k0, lane);
      acc[t] = wmma_f16(a, b, acc[t]);
    }
  }

  const int nl = lane & 15, hi = lane >> 4;
#pragma unroll
  for (int t = 0; t < 8; ++t) {
    const int n = tileN + t * 16 + nl;
    const float bv = bias[n];
#pragma unroll
    for (int r = 0; r < 8; ++r) {
      const int m = tileM + r + hi * 8;
      const float v = acc[t][r] + bv;
      if (OUT_F32) {
        outF[(size_t)m * D_MODEL + n] = v;
      } else if (HEAD_SPLIT) {
        const int b = m >> 11;        // m / SEQ
        const int s = m & (SEQ - 1);  // m % SEQ
        const int h = n >> 6;         // n / DK
        const int d = n & 63;         // n % DK
        outH[(((size_t)(b * NH + h) * SEQ) + s) * DK + d] = (_Float16)v;
      } else {
        outH[(size_t)m * D_MODEL + n] = (_Float16)v;
      }
    }
  }
}

// ---------------------------------------------------------------------------
// Flash attention. Qp/Kp/Vp are f16 [B*H, S, DK]. X output f16 [B*S, D_MODEL].
// Block = 256 threads = 8 waves; each wave owns 16 query rows (block: 128).
// Grid = (SEQ/128, B*NH). Key blocks of 64: double-buffered TDM DMA into LDS
// (hardware row padding -> 72-half pitch); V blocks staged transposed.
// ---------------------------------------------------------------------------
__global__ __launch_bounds__(256) void flash_attn(const _Float16* __restrict__ Qp,
                                                  const _Float16* __restrict__ Kp,
                                                  const _Float16* __restrict__ Vp,
                                                  _Float16* __restrict__ X) {
  constexpr int LDK = DK + 8;  // 72 halves per K row (matches TDM pad config)
  constexpr int LDV = 64 + 8;  // 72 halves per Vt row
  constexpr int LDP = 64 + 8;  // 72 halves per P row

  __shared__ _Float16 Klds[2][64 * LDK];    // double-buffered [key][d]
  __shared__ _Float16 Vt[DK * LDV];         // [d][key]  (transposed V)
  __shared__ _Float16 Plds[8 * 16 * LDP];   // per-wave [16][64] probs

  const int lane = threadIdx.x & 31;
  const int wave = threadIdx.x >> 5;
  const int nl = lane & 15, hi = lane >> 4;
  const int bh = blockIdx.y;                       // b*NH + h
  const int q0 = blockIdx.x * 128 + wave * 16;

  const _Float16* Qb = Qp + (size_t)bh * SEQ * DK;
  const _Float16* Kb = Kp + (size_t)bh * SEQ * DK;
  const _Float16* Vb = Vp + (size_t)bh * SEQ * DK;

  // Persistent Q fragments: k-steps over DK (0..31, 32..63)
  v16h qf0 = load_a_frag(Qb, DK, q0, 0, lane);
  v16h qf1 = load_a_frag(Qb, DK, q0, 32, lane);

  v8f o[4] = {};
  float mrow[8], lrow[8];
#pragma unroll
  for (int r = 0; r < 8; ++r) { mrow[r] = -__builtin_inff(); lrow[r] = 0.0f; }

  const float sc = 0.125f;  // 1/sqrt(DK)
  _Float16* Pw = &Plds[wave * 16 * LDP];

#if USE_TDM
  if (wave == 0) {
    tdm_load_kv_tile(Kb, (unsigned)(uintptr_t)&Klds[0][0], SEQ);
  }
#endif

  int buf = 0;
  for (int j0 = 0; j0 < SEQ; j0 += 64) {
    // ---- stage V block transposed (all threads) ----
    {
      const int t = threadIdx.x;      // 0..255
      const int key = t >> 2;         // 0..63
      const int dbase = (t & 3) * 16; // 0,16,32,48
      const _Float16* vg = Vb + (size_t)(j0 + key) * DK + dbase;
      v8h v0 = *(const v8h*)(vg);
      v8h v1 = *(const v8h*)(vg + 8);
#pragma unroll
      for (int i = 0; i < 8; ++i) {
        Vt[(dbase + i) * LDV + key] = v0[i];
        Vt[(dbase + 8 + i) * LDV + key] = v1[i];
      }
    }

    // ---- K block: TDM double-buffer (or manual fallback) ----
#if USE_TDM
    if (wave == 0) {
      if (j0 + 64 < SEQ) {
        // prefetch next K tile into the other buffer, then wait for current
        tdm_load_kv_tile(Kb + (size_t)(j0 + 64) * DK,
                         (unsigned)(uintptr_t)&Klds[buf ^ 1][0], SEQ - j0 - 64);
        __builtin_amdgcn_s_wait_tensorcnt(1);
      } else {
        __builtin_amdgcn_s_wait_tensorcnt(0);
      }
    }
#else
    {
      const int t = threadIdx.x;
      const int key = t >> 2;
      const int dbase = (t & 3) * 16;
      const _Float16* kg = Kb + (size_t)(j0 + key) * DK + dbase;
      v8h k0v = *(const v8h*)(kg);
      v8h k1v = *(const v8h*)(kg + 8);
      *(v8h*)(&Klds[buf][key * LDK + dbase]) = k0v;
      *(v8h*)(&Klds[buf][key * LDK + dbase + 8]) = k1v;
    }
#endif
    __syncthreads();

    // ---- scores S = Q (16xDK) . K_block^T  -> 16x64, scaled ----
    v8f st[4];
#pragma unroll
    for (int jt = 0; jt < 4; ++jt) {
      v8f acc = {};
      const _Float16* kp = &Klds[buf][(jt * 16 + nl) * LDK + hi * 8];
      frag16 ub;
      ub.h[0] = *(const v8h*)(kp);
      ub.h[1] = *(const v8h*)(kp + 16);
      acc = wmma_f16(qf0, ub.v, acc);
      ub.h[0] = *(const v8h*)(kp + 32);
      ub.h[1] = *(const v8h*)(kp + 48);
      acc = wmma_f16(qf1, ub.v, acc);
#pragma unroll
      for (int r = 0; r < 8; ++r) acc[r] *= sc;
      st[jt] = acc;
    }

    // ---- online softmax: row max, rescale running state ----
#pragma unroll
    for (int r = 0; r < 8; ++r) {
      float mx = fmaxf(fmaxf(st[0][r], st[1][r]), fmaxf(st[2][r], st[3][r]));
      mx = fmaxf(mx, __shfl_xor(mx, 1, 32));
      mx = fmaxf(mx, __shfl_xor(mx, 2, 32));
      mx = fmaxf(mx, __shfl_xor(mx, 4, 32));
      mx = fmaxf(mx, __shfl_xor(mx, 8, 32));
      const float mnew = fmaxf(mrow[r], mx);
      const float resc = __expf(mrow[r] - mnew);  // exp(-inf)=0 on first block
      mrow[r] = mnew;
      lrow[r] *= resc;
#pragma unroll
      for (int t = 0; t < 4; ++t) o[t][r] *= resc;
    }

    // ---- P = exp(S - m), f16 into per-wave LDS; accumulate row sums ----
    float rsum[8] = {0.f, 0.f, 0.f, 0.f, 0.f, 0.f, 0.f, 0.f};
#pragma unroll
    for (int jt = 0; jt < 4; ++jt) {
#pragma unroll
      for (int r = 0; r < 8; ++r) {
        const float p = __expf(st[jt][r] - mrow[r]);
        rsum[r] += p;
        Pw[(r + hi * 8) * LDP + jt * 16 + nl] = (_Float16)p;
      }
    }
#pragma unroll
    for (int r = 0; r < 8; ++r) {
      float s = rsum[r];
      s += __shfl_xor(s, 1, 32);
      s += __shfl_xor(s, 2, 32);
      s += __shfl_xor(s, 4, 32);
      s += __shfl_xor(s, 8, 32);
      lrow[r] += s;
    }

    // ---- O += P (16x64) . V_block (64xDK) ----
#pragma unroll
    for (int ks = 0; ks < 2; ++ks) {
      const _Float16* pp = Pw + nl * LDP + ks * 32 + hi * 8;  // A-frag: row=nl, ksel=hi
      frag16 ua;
      ua.h[0] = *(const v8h*)(pp);
      ua.h[1] = *(const v8h*)(pp + 16);
#pragma unroll
      for (int t = 0; t < 4; ++t) {
        const _Float16* vp = &Vt[(t * 16 + nl) * LDV + ks * 32 + hi * 8];
        frag16 ub;
        ub.h[0] = *(const v8h*)(vp);
        ub.h[1] = *(const v8h*)(vp + 16);
        o[t] = wmma_f16(ua.v, ub.v, o[t]);
      }
    }
    __syncthreads();
    buf ^= 1;
  }

  // ---- normalize and write X[b*SEQ + q, h*DK + d] ----
  const int b = bh >> 4;
  const int h = bh & 15;
#pragma unroll
  for (int r = 0; r < 8; ++r) {
    const float inv = 1.0f / lrow[r];
#pragma unroll
    for (int t = 0; t < 4; ++t) o[t][r] *= inv;
  }
#pragma unroll
  for (int t = 0; t < 4; ++t) {
    const int d = t * 16 + nl;
#pragma unroll
    for (int r = 0; r < 8; ++r) {
      const int qr = q0 + r + hi * 8;
      X[((size_t)(b * SEQ + qr)) * D_MODEL + h * DK + d] = (_Float16)o[t][r];
    }
  }
}

// ---------------------------------------------------------------------------
// Launcher. Inputs: q,k,v, w_q,b_q, w_k,b_k, w_v,b_v, w_o,b_o (all f32).
// Workspace layout (f16): qf,kf,vf [4096x1024], wq,wk,wv,wo [1024x1024],
// Qp,Kp,Vp [B*H,S,DK], X [4096x1024]  -> exactly 64 MiB.
// ---------------------------------------------------------------------------
extern "C" void kernel_launch(void* const* d_in, const int* in_sizes, int n_in,
                              void* d_out, int out_size, void* d_ws, size_t ws_size,
                              hipStream_t stream) {
  (void)in_sizes; (void)n_in; (void)out_size; (void)ws_size;

  const float* q   = (const float*)d_in[0];
  const float* k   = (const float*)d_in[1];
  const float* v   = (const float*)d_in[2];
  const float* w_q = (const float*)d_in[3];
  const float* b_q = (const float*)d_in[4];
  const float* w_k = (const float*)d_in[5];
  const float* b_k = (const float*)d_in[6];
  const float* w_v = (const float*)d_in[7];
  const float* b_v = (const float*)d_in[8];
  const float* w_o = (const float*)d_in[9];
  const float* b_o = (const float*)d_in[10];
  float* out = (float*)d_out;

  const size_t NTOK = (size_t)MROWS * D_MODEL;       // 4,194,304
  const size_t NWGT = (size_t)D_MODEL * D_MODEL;     // 1,048,576

  _Float16* p  = (_Float16*)d_ws;
  _Float16* qf = p; p += NTOK;
  _Float16* kf = p; p += NTOK;
  _Float16* vf = p; p += NTOK;
  _Float16* wq = p; p += NWGT;
  _Float16* wk = p; p += NWGT;
  _Float16* wv = p; p += NWGT;
  _Float16* wo = p; p += NWGT;
  _Float16* Qp = p; p += NTOK;
  _Float16* Kp = p; p += NTOK;
  _Float16* Vp = p; p += NTOK;
  _Float16* X  = p; p += NTOK;

  // 1) casts to f16
  cast_f32_f16<<<dim3((unsigned)(NTOK / 1024)), dim3(256), 0, stream>>>(q, qf, (int)NTOK);
  cast_f32_f16<<<dim3((unsigned)(NTOK / 1024)), dim3(256), 0, stream>>>(k, kf, (int)NTOK);
  cast_f32_f16<<<dim3((unsigned)(NTOK / 1024)), dim3(256), 0, stream>>>(v, vf, (int)NTOK);
  cast_f32_f16<<<dim3((unsigned)(NWGT / 1024)), dim3(256), 0, stream>>>(w_q, wq, (int)NWGT);
  cast_f32_f16<<<dim3((unsigned)(NWGT / 1024)), dim3(256), 0, stream>>>(w_k, wk, (int)NWGT);
  cast_f32_f16<<<dim3((unsigned)(NWGT / 1024)), dim3(256), 0, stream>>>(w_v, wv, (int)NWGT);
  cast_f32_f16<<<dim3((unsigned)(NWGT / 1024)), dim3(256), 0, stream>>>(w_o, wo, (int)NWGT);

  // 2) QKV projections: [4096,1024] @ [1024,1024]^T -> head-split f16
  dim3 ggrid(D_MODEL / 512, MROWS / 32);  // (2, 128)
  dim3 gblk(256);
  gemm_xwT<false, true><<<ggrid, gblk, 0, stream>>>(qf, wq, b_q, nullptr, Qp, D_MODEL);
  gemm_xwT<false, true><<<ggrid, gblk, 0, stream>>>(kf, wk, b_k, nullptr, Kp, D_MODEL);
  gemm_xwT<false, true><<<ggrid, gblk, 0, stream>>>(vf, wv, b_v, nullptr, Vp, D_MODEL);

  // 3) flash attention
  flash_attn<<<dim3(SEQ / 128, BATCH * NH), dim3(256), 0, stream>>>(Qp, Kp, Vp, X);

  // 4) output projection: X @ w_o^T + b_o -> f32 d_out
  gemm_xwT<true, false><<<ggrid, gblk, 0, stream>>>(X, wo, b_o, out, nullptr, D_MODEL);
}
